// NonGraph_Model_62947040690539
// MI455X (gfx1250) — matrix operand order repaired and checked by reading, not compiled, not run on hardware
//
#include <hip/hip_runtime.h>
#include <math.h>

typedef __attribute__((ext_vector_type(16))) _Float16 v16h;
typedef __attribute__((ext_vector_type(8)))  _Float16 v8h;
typedef __attribute__((ext_vector_type(8)))  float    v8f;
typedef __attribute__((ext_vector_type(4)))  float    v4f;

#define DEV static __device__ __forceinline__

constexpr int kN = 200000;
constexpr int kF = 512;
constexpr int kRowsPerWG = 128;
constexpr unsigned kTotalZ = 1600000u;      // N*A
constexpr unsigned kHalfZ  = kTotalZ / 2;   // threefry counter split point

DEV v8f wmma16(v16h a, v16h b, v8f c) {
  // v_wmma_f32_16x16x32_f16 (probe-confirmed 8-arg builtin)
  return __builtin_amdgcn_wmma_f32_16x16x32_f16(false, a, false, b, (short)0, c, false, false);
}

DEV unsigned rotl32(unsigned x, unsigned n){ return (x << n) | (x >> (32u - n)); }

// Reproduce jax.random.normal(key(42), (N,8))[j] : threefry2x32-20 -> uniform(-1,1) -> sqrt(2)*erfinv
DEV float jax_normal(unsigned j){
  unsigned c0, c1; bool use0;
  if (j < kHalfZ){ c0 = j; c1 = j + kHalfZ; use0 = true; }
  else           { c0 = j - kHalfZ; c1 = j; use0 = false; }
  const unsigned ks0 = 0u, ks1 = 42u, ks2 = ks0 ^ ks1 ^ 0x1BD11BDAu;
  const unsigned ks[3] = { ks0, ks1, ks2 };
  const unsigned R[8] = { 13u,15u,26u,6u,17u,29u,16u,24u };
  unsigned x0 = c0 + ks0, x1 = c1 + ks1;
  int inj = 0;
  #pragma unroll
  for (int i = 0; i < 20; ++i){
    x0 += x1; x1 = rotl32(x1, R[i & 7]); x1 ^= x0;
    if ((i & 3) == 3){ ++inj; x0 += ks[inj % 3]; x1 += ks[(inj + 1) % 3] + (unsigned)inj; }
  }
  const unsigned bits = use0 ? x0 : x1;
  const float f = __uint_as_float((bits >> 9) | 0x3F800000u) - 1.0f;   // [0,1)
  const float lo = -0.99999994f, hi = 1.0f;
  const float u = fmaxf(lo, f * (hi - lo) + lo);
  return 1.4142135623730951f * erfinvf(u);
}

__global__ __launch_bounds__(256) void naf_fused(
    const float* __restrict__ X, const float* __restrict__ RL,
    const float* __restrict__ action,
    const float* __restrict__ W1, const float* __restrict__ b1,
    const float* __restrict__ W2, const float* __restrict__ b2,
    const float* __restrict__ Wv, const float* __restrict__ bv,
    const float* __restrict__ Wmu, const float* __restrict__ bmu,
    const float* __restrict__ WL, const float* __restrict__ bL,
    float* __restrict__ oSamp, float* __restrict__ oQ, float* __restrict__ oV)
{
  __shared__ __align__(32) _Float16 w1s[16*2*32*16];  // 32 KB: B-frag swizzled W1 (f16)
  __shared__ __align__(32) _Float16 w2s[2*32*16];     //  2 KB
  __shared__ __align__(32) _Float16 whs[2*32*16];     //  2 KB: [mu(8)|Ldiag(8)|v(1)|pad]
  __shared__ __align__(16) _Float16 hbuf[8][16*32];   //  8 KB: per-wave activation relayout
  __shared__ float obuf[8][16*32];                    // 16 KB: per-wave head outputs
  __shared__ float lb1[32], lb2[32], lbh[32];

  const int tid  = threadIdx.x;
  const int lane = tid & 31;
  const int wave = tid >> 5;
  const int col  = lane & 15;   // tile column (B/C) == tile row M (A)
  const int hiH  = lane >> 4;   // 0: lanes 0-15, 1: lanes 16-31

  // ---- stage biases ----
  if (tid < 32){
    lb1[tid] = b1[tid];
    lb2[tid] = b2[tid];
    float hb = 0.0f;
    if (tid < 8)        hb = bmu[tid];
    else if (tid < 16){ const int i = tid - 8; hb = bL[(i*(i+3)) >> 1]; }
    else if (tid == 16) hb = bv[0];
    lbh[tid] = hb;
  }
  // ---- stage W1 into B-fragment lane order: chunk q=(kb*2+t)*32+ln, 16 f16 each ----
  for (int q = tid; q < 16*2*32; q += 256){
    const int kb = q >> 6, t = (q >> 5) & 1, ln = q & 31;
    const int c0 = t*16 + (ln & 15);
    const int kbase = kb*32 + ((ln >= 16) ? 8 : 0);
    _Float16* dst = &w1s[q*16];
    #pragma unroll
    for (int c = 0; c < 16; ++c) dst[c] = (_Float16)W1[(kbase + c + ((c >= 8) ? 8 : 0))*32 + c0];
  }
  // ---- stage W2 ----
  for (int q = tid; q < 2*32; q += 256){
    const int t = q >> 5, ln = q & 31;
    const int c0 = t*16 + (ln & 15);
    const int kbase = (ln >= 16) ? 8 : 0;
    _Float16* dst = &w2s[q*16];
    #pragma unroll
    for (int c = 0; c < 16; ++c) dst[c] = (_Float16)W2[(kbase + c + ((c >= 8) ? 8 : 0))*32 + c0];
  }
  // ---- stage head weights (P is diagonal -> only 8 of 36 WL columns matter) ----
  for (int q = tid; q < 2*32; q += 256){
    const int t = q >> 5, ln = q & 31;
    const int c0 = t*16 + (ln & 15);
    const int kbase = (ln >= 16) ? 8 : 0;
    _Float16* dst = &whs[q*16];
    #pragma unroll
    for (int c = 0; c < 16; ++c){
      const int K = kbase + c + ((c >= 8) ? 8 : 0);
      float v = 0.0f;
      if (c0 < 8)        v = Wmu[K*8 + c0];
      else if (c0 < 16){ const int i = c0 - 8; v = WL[K*36 + ((i*(i+3)) >> 1)]; }
      else if (c0 == 16) v = Wv[K];
      dst[c] = (_Float16)v;
    }
  }
  __syncthreads();

  const int rowBase = blockIdx.x * kRowsPerWG + wave * 16;
  const int row = rowBase + col;
  int lrow = row; if (lrow > kN - 1) lrow = kN - 1;    // clamp: WMMA needs EXEC all-ones
  const float* xrow = X + (size_t)lrow * kF + (hiH ? 8 : 0);

  const v16h* w1v = (const v16h*)w1s;
  const v16h* w2v = (const v16h*)w2s;
  const v16h* whv = (const v16h*)whs;

  // ---- layer 1: h1 = relu(X @ W1 + b1), 16 rows x 32 cols / wave ----
  v8f acc0, acc1;
  {
    const float bb0 = lb1[col], bb1 = lb1[16 + col];
    #pragma unroll
    for (int r = 0; r < 8; ++r){ acc0[r] = bb0; acc1[r] = bb1; }
  }
  for (int kb = 0; kb < 16; ++kb){
    const float* p = xrow + kb*32;
    const v4f a0 = __builtin_nontemporal_load((const v4f*)(p + 0));
    const v4f a1 = __builtin_nontemporal_load((const v4f*)(p + 4));
    const v4f a2 = __builtin_nontemporal_load((const v4f*)(p + 16));
    const v4f a3 = __builtin_nontemporal_load((const v4f*)(p + 20));
    v16h af;
    #pragma unroll
    for (int i = 0; i < 4; ++i){
      af[i]      = (_Float16)a0[i];
      af[i + 4]  = (_Float16)a1[i];
      af[i + 8]  = (_Float16)a2[i];
      af[i + 12] = (_Float16)a3[i];
    }
    acc0 = wmma16(af, w1v[(kb*2 + 0)*32 + lane], acc0);
    acc1 = wmma16(af, w1v[(kb*2 + 1)*32 + lane], acc1);
  }
  // relu -> hbuf[M][K] (f16) so it can be reloaded as an A fragment
  _Float16* hb = hbuf[wave];
  #pragma unroll
  for (int r = 0; r < 8; ++r){
    const int M = r + 8*hiH;
    hb[M*32 + col]      = (_Float16)fmaxf(acc0[r], 0.0f);
    hb[M*32 + 16 + col] = (_Float16)fmaxf(acc1[r], 0.0f);
  }
  asm volatile("s_wait_dscnt 0x0" ::: "memory");

  // ---- layer 2: h2 = relu(h1 @ W2 + b2) ----
  const int klo = hiH ? 8 : 0;
  v16h af2;
  {
    const v8h pa = *(const v8h*)&hb[col*32 + klo];
    const v8h pb = *(const v8h*)&hb[col*32 + klo + 16];
    #pragma unroll
    for (int i = 0; i < 8; ++i){ af2[i] = pa[i]; af2[i + 8] = pb[i]; }
  }
  v8f acc2a, acc2b;
  {
    const float bb0 = lb2[col], bb1 = lb2[16 + col];
    #pragma unroll
    for (int r = 0; r < 8; ++r){ acc2a[r] = bb0; acc2b[r] = bb1; }
  }
  acc2a = wmma16(af2, w2v[lane],      acc2a);
  acc2b = wmma16(af2, w2v[32 + lane], acc2b);
  #pragma unroll
  for (int r = 0; r < 8; ++r){
    const int M = r + 8*hiH;
    hb[M*32 + col]      = (_Float16)fmaxf(acc2a[r], 0.0f);
    hb[M*32 + 16 + col] = (_Float16)fmaxf(acc2b[r], 0.0f);
  }
  asm volatile("s_wait_dscnt 0x0" ::: "memory");

  // ---- heads: 16x32 tile = [mu(0:8) | Ldiag(8:16) | value(16) | zero pad] ----
  v16h af3;
  {
    const v8h pa = *(const v8h*)&hb[col*32 + klo];
    const v8h pb = *(const v8h*)&hb[col*32 + klo + 16];
    #pragma unroll
    for (int i = 0; i < 8; ++i){ af3[i] = pa[i]; af3[i + 8] = pb[i]; }
  }
  v8f acch0, acch1;
  {
    const float bb0 = lbh[col], bb1 = lbh[16 + col];
    #pragma unroll
    for (int r = 0; r < 8; ++r){ acch0[r] = bb0; acch1[r] = bb1; }
  }
  acch0 = wmma16(af3, whv[lane],      acch0);
  acch1 = wmma16(af3, whv[32 + lane], acch1);

  float* ob = obuf[wave];
  #pragma unroll
  for (int r = 0; r < 8; ++r){
    const int M = r + 8*hiH;
    ob[M*32 + col]      = acch0[r];
    ob[M*32 + 16 + col] = acch1[r];
  }
  asm volatile("s_wait_dscnt 0x0" ::: "memory");

  // ---- epilogue: lane pair (l, l+16) handles row `col`, 4 action dims each ----
  const bool valid = row < kN;
  const float val = ob[col*32 + 16];
  const float act = action[lrow];
  const float rli = RL[lrow];
  float ssum = 0.0f;
  float sp[4];
  #pragma unroll
  for (int c = 0; c < 4; ++c){
    const int a = hiH*4 + c;
    const float mu = tanhf(ob[col*32 + a]);
    const float d  = expf(tanhf(ob[col*32 + 8 + a]));   // L diag; P_kk = d^2, Lc_kk = 1/d
    const float am = act - mu;
    ssum += am*am*d*d;
    const float z = jax_normal((unsigned)row*8u + (unsigned)a);
    float s = mu + z / d;
    s = fminf(1.0f, fmaxf(-1.0f, s)) * rli;
    sp[c] = s;
  }
  const float ssum_other = __shfl_xor(ssum, 16);
  const float Adv = -0.5f * (ssum + ssum_other);
  if (valid){
    const v4f sv = { sp[0], sp[1], sp[2], sp[3] };
    __builtin_nontemporal_store(sv, (v4f*)&oSamp[(size_t)row*8 + hiH*4]);
    if (!hiH){
      __builtin_nontemporal_store(Adv + val, &oQ[row]);
      __builtin_nontemporal_store(val, &oV[row]);
    }
  }
}

extern "C" void kernel_launch(void* const* d_in, const int* in_sizes, int n_in,
                              void* d_out, int out_size, void* d_ws, size_t ws_size,
                              hipStream_t stream) {
  (void)in_sizes; (void)n_in; (void)out_size; (void)d_ws; (void)ws_size;
  const float* X   = (const float*)d_in[0];
  const float* RL  = (const float*)d_in[1];
  const float* act = (const float*)d_in[2];
  const float* W1  = (const float*)d_in[3];
  const float* b1  = (const float*)d_in[4];
  const float* W2  = (const float*)d_in[5];
  const float* b2  = (const float*)d_in[6];
  const float* Wv  = (const float*)d_in[7];
  const float* bv  = (const float*)d_in[8];
  const float* Wmu = (const float*)d_in[9];
  const float* bmu = (const float*)d_in[10];
  const float* WL  = (const float*)d_in[11];
  const float* bL  = (const float*)d_in[12];
  float* out   = (float*)d_out;
  float* oSamp = out;                       // [N,8]
  float* oQ    = out + (size_t)kN * 8;      // [N,1]
  float* oV    = out + (size_t)kN * 9;      // [N,1]
  dim3 grid((kN + kRowsPerWG - 1) / kRowsPerWG);
  naf_fused<<<grid, 256, 0, stream>>>(X, RL, act, W1, b1, W2, b2, Wv, bv,
                                      Wmu, bmu, WL, bL, oSamp, oQ, oV);
}
